// CausalSelfAttention_74938589380902
// MI455X (gfx1250) — matrix operand order repaired and checked by reading, not compiled, not run on hardware
//
#include <hip/hip_runtime.h>

// ---------------------------------------------------------------------------
// Problem constants (from the reference)
// ---------------------------------------------------------------------------
#define D_MODEL   1024
#define N_HEADS   16
#define HEAD_DIM  64
#define BATCH     4
#define SEQ       1024
#define M_TOTAL   (BATCH * SEQ)          // 4096 rows of x
#define EPS       1e-6f
#define SM_SCALE  0.125f                  // 1/sqrt(64)

typedef __attribute__((ext_vector_type(16))) __bf16 v16bf;
typedef __attribute__((ext_vector_type(8)))  float  v8f;

// ---------------------------------------------------------------------------
// WMMA helper: D = A(16x32 bf16) * B(32x16 bf16) + C(16x16 f32)
// ---------------------------------------------------------------------------
static __device__ __forceinline__ v8f wmma_bf16(v16bf a, v16bf b, v8f c) {
  return __builtin_amdgcn_wmma_f32_16x16x32_bf16(
      /*neg_a=*/false, a, /*neg_b=*/false, b,
      /*c_mod=*/(short)0, c, /*reuse_a=*/false, /*reuse_b=*/false);
}

// ---------------------------------------------------------------------------
// Async global->LDS 16B copy (GLOBAL_LOAD_ASYNC_TO_LDS_B128, ASYNCcnt).
// ASYNCcnt retires loads in order, so partial waits implement double buffering.
// ---------------------------------------------------------------------------
static __device__ __forceinline__ void async_g2l_b128(void* lds_dst,
                                                      const void* gsrc) {
  unsigned int       l = (unsigned int)(unsigned long long)(uintptr_t)lds_dst;
  unsigned long long g = (unsigned long long)(uintptr_t)gsrc;
  asm volatile("global_load_async_to_lds_b128 %0, %1, off"
               :: "v"(l), "v"(g) : "memory");
}
static __device__ __forceinline__ void async_wait_0() {
  asm volatile("s_wait_asynccnt 0x0" ::: "memory");
}
static __device__ __forceinline__ void async_wait_2() {
  asm volatile("s_wait_asynccnt 0x2" ::: "memory");
}
static __device__ __forceinline__ void async_wait_4() {
  asm volatile("s_wait_asynccnt 0x4" ::: "memory");
}

// ---------------------------------------------------------------------------
// Fragment loaders from LDS (contiguous k = 16*(lane/16) + s, same order for A and B)
// ---------------------------------------------------------------------------
// A-operand (16x32) from row-major [m][k], row stride ldm.
static __device__ __forceinline__
v16bf load_a_frag(const __bf16* lds, int ldm, int row0, int col0, int lane) {
  int m    = row0 + (lane & 15);
  int base = (lane >> 4) * 16;
  const __bf16* p = lds + m * ldm + col0 + base;
  v16bf f;
#pragma unroll
  for (int i = 0; i < 8; ++i)  f[i] = p[i];
#pragma unroll
  for (int i = 8; i < 16; ++i) f[i] = p[i];
  return f;
}

// B-operand (32x16) from an [n][k] row-major tile (row stride ldk):
// lane's 16 elements are 32 contiguous bytes -> 2x ds_load_b128.
static __device__ __forceinline__
v16bf load_b_frag_nk(const __bf16* lds, int ldk, int n0, int k0, int lane) {
  int n  = n0 + (lane & 15);
  int kb = k0 + (lane >> 4) * 16;
  const __bf16* p = lds + n * ldk + kb;
  v16bf f;
#pragma unroll
  for (int i = 0; i < 16; ++i) f[i] = p[i];
  return f;
}

// Deterministic sum across the 16 lanes sharing a C-matrix row group.
static __device__ __forceinline__ float rowred16(float v) {
  v += __shfl_xor(v, 1, 32);
  v += __shfl_xor(v, 2, 32);
  v += __shfl_xor(v, 4, 32);
  v += __shfl_xor(v, 8, 32);
  return v;
}

// ---------------------------------------------------------------------------
// Kernel 1: f32 -> bf16 conversion
// ---------------------------------------------------------------------------
__global__ void cvt_f32_bf16(const float* __restrict__ in,
                             __bf16* __restrict__ out, int n) {
  int i = blockIdx.x * blockDim.x + threadIdx.x;
  if (i < n) out[i] = (__bf16)in[i];
}

// ---------------------------------------------------------------------------
// Kernel 2: fused qkv + k_y projection (double-buffered async pipeline).
//   A = x_bf [4096 x 1024]; virtual N = 4096 (3072 qkv rows, 1024 ky rows).
//   q/k/ky written head-split [B,H,T,64]; V written TRANSPOSED [B,H,64,T].
// ---------------------------------------------------------------------------
#define TLD 72           // [64][64] tile row stride (+pad, 16B-aligned rows)
#define NKT (D_MODEL / 64)

__global__ __launch_bounds__(256)
void gemm_qkv_ky(const __bf16* __restrict__ xb,
                 const __bf16* __restrict__ wqkv,
                 const __bf16* __restrict__ wky,
                 __bf16* __restrict__ qo, __bf16* __restrict__ ko,
                 __bf16* __restrict__ vo /* [B,H,hd,T] */,
                 __bf16* __restrict__ kyo) {
  __shared__ __bf16 s_a[2][64 * TLD];
  __shared__ __bf16 s_b[2][64 * TLD];

  const int mt = blockIdx.x, nt = blockIdx.y;
  const int tid = threadIdx.x, lane = tid & 31, wv = tid >> 5;
  const int ni = wv & 3, mi0 = (wv >> 2);

  const int n0g = nt * 64;
  const __bf16* wmat = (n0g < 3 * D_MODEL) ? wqkv : wky;
  const int nrow0 = (n0g < 3 * D_MODEL) ? n0g : (n0g - 3 * D_MODEL);

  // Stage one 64x64 A tile + 64x64 B tile: 4 async b128 per thread.
  auto stage = [&](int kt, int pb) {
#pragma unroll
    for (int c = 0; c < 2; ++c) {
      int ch = tid + c * 256;
      int r = ch >> 3, k0 = (ch & 7) * 8;
      async_g2l_b128(&s_a[pb][r * TLD + k0],
                     xb + (long)(mt * 64 + r) * D_MODEL + kt * 64 + k0);
      async_g2l_b128(&s_b[pb][r * TLD + k0],
                     wmat + (long)(nrow0 + r) * D_MODEL + kt * 64 + k0);
    }
  };

  v8f acc0 = {}, acc1 = {};
  int p = 0;
  stage(0, 0);
  for (int kt = 0; kt < NKT; ++kt) {
    __syncthreads();                       // readers of buf[1-p] are done
    bool pre = (kt + 1) < NKT;
    if (pre) stage(kt + 1, p ^ 1);         // prefetch next tile
    if (pre) async_wait_4(); else async_wait_0();   // tile kt resident
    __syncthreads();
    const __bf16* pa = s_a[p];
    const __bf16* pb = s_b[p];
    v16bf b0  = load_b_frag_nk(pb, TLD, ni * 16, 0, lane);
    v16bf b1  = load_b_frag_nk(pb, TLD, ni * 16, 32, lane);
    v16bf a00 = load_a_frag(pa, TLD, mi0 * 16, 0, lane);
    v16bf a01 = load_a_frag(pa, TLD, mi0 * 16, 32, lane);
    v16bf a10 = load_a_frag(pa, TLD, (mi0 + 2) * 16, 0, lane);
    v16bf a11 = load_a_frag(pa, TLD, (mi0 + 2) * 16, 32, lane);
    acc0 = wmma_bf16(a00, b0, acc0);
    acc0 = wmma_bf16(a01, b1, acc0);
    acc1 = wmma_bf16(a10, b0, acc1);
    acc1 = wmma_bf16(a11, b1, acc1);
    p ^= 1;
  }

  // Scatter-store into head-split layout.
  const int n_g = n0g + ni * 16 + (lane & 15);
  __bf16* dst;
  int h, d;
  bool is_v = false;
  if (n_g < 3 * D_MODEL) {
    int which  = n_g >> 10;
    int within = n_g & (D_MODEL - 1);
    h = within >> 6; d = within & 63;
    dst = (which == 0) ? qo : (which == 1) ? ko : vo;
    is_v = (which == 2);
  } else {
    int f = n_g - 3 * D_MODEL;
    h = f >> 6; d = f & 63;
    dst = kyo;
  }
#pragma unroll
  for (int j = 0; j < 8; ++j) {
    int m_g0 = mt * 64 + mi0 * 16 + j + 8 * (lane >> 4);
    int m_g1 = m_g0 + 32;
    int b0i = m_g0 >> 10, t0 = m_g0 & (SEQ - 1);
    int b1i = m_g1 >> 10, t1 = m_g1 & (SEQ - 1);
    long i0, i1;
    if (is_v) {  // transposed value layout [B,H,hd,T]
      i0 = (((long)b0i * N_HEADS + h) * HEAD_DIM + d) * SEQ + t0;
      i1 = (((long)b1i * N_HEADS + h) * HEAD_DIM + d) * SEQ + t1;
    } else {     // [B,H,T,hd]
      i0 = (((long)b0i * N_HEADS + h) * SEQ + t0) * HEAD_DIM + d;
      i1 = (((long)b1i * N_HEADS + h) * SEQ + t1) * HEAD_DIM + d;
    }
    dst[i0] = (__bf16)acc0[j];
    dst[i1] = (__bf16)acc1[j];
  }
}

// ---------------------------------------------------------------------------
// Kernel 3: fused eml-norm attention, one (b, h, 32-row q-block) per WG.
//   Pass 1: agg_m = sum_{k<=m} exp(clip(scale * q@ky^T))
//   Pass 2: w = log1p((|scale*q@k^T|+eps)/(agg+2eps)) masked;
//           O = w @ v, rowsum = sum(w); out = O / (rowsum + eps)
// All staging is double-buffered async b128 (V arrives pre-transposed).
// ---------------------------------------------------------------------------
#define QLD 72
#define KLD 72
#define VLD 72
#define WLD 72

__global__ __launch_bounds__(256)
void attn_eml(const __bf16* __restrict__ qb, const __bf16* __restrict__ kb,
              const __bf16* __restrict__ vtb /* [B,H,hd,T] */,
              const __bf16* __restrict__ kyb,
              __bf16* __restrict__ attn_out) {
  __shared__ __bf16 s_q[32 * QLD];
  __shared__ __bf16 s_k[2][64 * KLD];
  __shared__ __bf16 s_vt[2][64 * VLD];
  __shared__ __bf16 s_w[32 * WLD];
  __shared__ float  s_part[4][32];
  __shared__ float  s_agg[32];
  __shared__ float  s_rsum[32];

  const int qblk = blockIdx.x;
  const int h    = blockIdx.y;
  const int b    = blockIdx.z;
  const int tid  = threadIdx.x, lane = tid & 31, wv = tid >> 5;
  const int mi = wv >> 2;                      // 0..1 (M sub-tile)
  const int ni = wv & 3;                       // 0..3 (N sub-tile)
  const int q0 = qblk * 32;
  const int ktiles = (q0 >> 6) + 1;            // causal: only lower k tiles
  const long bh  = ((long)b * N_HEADS + h) * SEQ;          // rows of q/k/ky
  const long bhd = ((long)b * N_HEADS + h) * HEAD_DIM;     // rows of v^T

  // Stage q block (32 x 64): one async b128 per thread.
  async_g2l_b128(&s_q[(tid >> 3) * QLD + (tid & 7) * 8],
                 qb + (bh + q0 + (tid >> 3)) * HEAD_DIM + (tid & 7) * 8);
  async_wait_0();
  __syncthreads();
  const v16bf aq0 = load_a_frag(s_q, QLD, mi * 16, 0, lane);
  const v16bf aq1 = load_a_frag(s_q, QLD, mi * 16, 32, lane);

  // ky tile stage: 2 async per thread. k+v tile stage: 4 async per thread.
  auto stage_ky = [&](int kt, int pb) {
#pragma unroll
    for (int c = 0; c < 2; ++c) {
      int ch = tid + c * 256;
      int r = ch >> 3, c0 = (ch & 7) * 8;
      async_g2l_b128(&s_k[pb][r * KLD + c0],
                     kyb + (bh + kt * 64 + r) * HEAD_DIM + c0);
    }
  };
  auto stage_kv = [&](int kt, int pb) {
#pragma unroll
    for (int c = 0; c < 2; ++c) {
      int ch = tid + c * 256;
      int r = ch >> 3, c0 = (ch & 7) * 8;
      async_g2l_b128(&s_k[pb][r * KLD + c0],
                     kb + (bh + kt * 64 + r) * HEAD_DIM + c0);      // K[kr][d]
      async_g2l_b128(&s_vt[pb][r * VLD + c0],
                     vtb + (bhd + r) * SEQ + kt * 64 + c0);         // V^T[d][kr]
    }
  };

  float aggp[8];
#pragma unroll
  for (int j = 0; j < 8; ++j) aggp[j] = 0.0f;

  // -------- Pass 1: denominator agg from q @ ky^T --------
  int p = 0;
  stage_ky(0, 0);
  for (int kt = 0; kt < ktiles; ++kt) {
    __syncthreads();
    bool pre = (kt + 1) < ktiles;
    if (pre) stage_ky(kt + 1, p ^ 1);
    if (pre) async_wait_2(); else async_wait_0();
    __syncthreads();
    v8f sy = {};
    sy = wmma_bf16(aq0, load_b_frag_nk(s_k[p], KLD, ni * 16, 0, lane), sy);
    sy = wmma_bf16(aq1, load_b_frag_nk(s_k[p], KLD, ni * 16, 32, lane), sy);
    const int colg = kt * 64 + ni * 16 + (lane & 15);
#pragma unroll
    for (int j = 0; j < 8; ++j) {
      int rowg = q0 + mi * 16 + j + 8 * (lane >> 4);
      if (colg <= rowg) {
        float v = fminf(fmaxf(sy[j] * SM_SCALE, -15.0f), 15.0f);
        aggp[j] += __expf(v);
      }
    }
    p ^= 1;
  }
  // Deterministic reduction: shuffle within 16-lane row groups, then fixed
  // order accumulation across the 4 ni waves.
#pragma unroll
  for (int j = 0; j < 8; ++j) {
    float s = rowred16(aggp[j]);
    if ((lane & 15) == 0) s_part[ni][mi * 16 + j + 8 * (lane >> 4)] = s;
  }
  __syncthreads();
  if (tid < 32)
    s_agg[tid] = s_part[0][tid] + s_part[1][tid] + s_part[2][tid] + s_part[3][tid];
  __syncthreads();

  float invagg[8];
#pragma unroll
  for (int j = 0; j < 8; ++j) {
    float a = s_agg[mi * 16 + j + 8 * (lane >> 4)];
    invagg[j] = 1.0f / (a + EPS + EPS);        // agg = sum+eps, then /(agg+eps)
  }

  // -------- Pass 2: scores -> w -> O = w @ v --------
  float rsump[8];
#pragma unroll
  for (int j = 0; j < 8; ++j) rsump[j] = 0.0f;
  v8f oacc = {};

  p = 0;
  stage_kv(0, 0);
  for (int kt = 0; kt < ktiles; ++kt) {
    __syncthreads();                      // prev iter: s_w + buf[1-p] reads done
    bool pre = (kt + 1) < ktiles;
    if (pre) stage_kv(kt + 1, p ^ 1);
    if (pre) async_wait_4(); else async_wait_0();
    __syncthreads();
    v8f sc = {};
    sc = wmma_bf16(aq0, load_b_frag_nk(s_k[p], KLD, ni * 16, 0, lane), sc);
    sc = wmma_bf16(aq1, load_b_frag_nk(s_k[p], KLD, ni * 16, 32, lane), sc);
    const int colg = kt * 64 + ni * 16 + (lane & 15);
#pragma unroll
    for (int j = 0; j < 8; ++j) {
      int mrow = mi * 16 + j + 8 * (lane >> 4);
      int rowg = q0 + mrow;
      float w = 0.0f;
      if (colg <= rowg) {
        float l = sc[j] * SM_SCALE;
        w = log1pf((fabsf(l) + EPS) * invagg[j]);   // softplus(eml ratio)
      }
      rsump[j] += w;
      s_w[mrow * WLD + ni * 16 + (lane & 15)] = (__bf16)w;
    }
    __syncthreads();
    oacc = wmma_bf16(load_a_frag(s_w, WLD, mi * 16, 0, lane),
                     load_b_frag_nk(s_vt[p], VLD, ni * 16, 0, lane), oacc);
    oacc = wmma_bf16(load_a_frag(s_w, WLD, mi * 16, 32, lane),
                     load_b_frag_nk(s_vt[p], VLD, ni * 16, 32, lane), oacc);
    p ^= 1;
  }

#pragma unroll
  for (int j = 0; j < 8; ++j) {
    float s = rowred16(rsump[j]);
    if ((lane & 15) == 0) s_part[ni][mi * 16 + j + 8 * (lane >> 4)] = s;
  }
  __syncthreads();
  if (tid < 32)
    s_rsum[tid] = s_part[0][tid] + s_part[1][tid] + s_part[2][tid] + s_part[3][tid];
  __syncthreads();

#pragma unroll
  for (int j = 0; j < 8; ++j) {
    int mrow = mi * 16 + j + 8 * (lane >> 4);
    float o = oacc[j] / (s_rsum[mrow] + EPS);
    int t = q0 + mrow;
    int dc = h * HEAD_DIM + ni * 16 + (lane & 15);
    attn_out[((long)b * SEQ + t) * D_MODEL + dc] = (__bf16)o;
  }
}

// ---------------------------------------------------------------------------
// Kernel 4: output projection, out = attn_bf @ w_proj^T -> f32 d_out
// (same double-buffered pipeline as kernel 2)
// ---------------------------------------------------------------------------
__global__ __launch_bounds__(256)
void gemm_proj(const __bf16* __restrict__ ab, const __bf16* __restrict__ wproj,
               float* __restrict__ out) {
  __shared__ __bf16 s_a[2][64 * TLD];
  __shared__ __bf16 s_b[2][64 * TLD];

  const int mt = blockIdx.x, nt = blockIdx.y;
  const int tid = threadIdx.x, lane = tid & 31, wv = tid >> 5;
  const int ni = wv & 3, mi0 = (wv >> 2);

  auto stage = [&](int kt, int pb) {
#pragma unroll
    for (int c = 0; c < 2; ++c) {
      int ch = tid + c * 256;
      int r = ch >> 3, k0 = (ch & 7) * 8;
      async_g2l_b128(&s_a[pb][r * TLD + k0],
                     ab + (long)(mt * 64 + r) * D_MODEL + kt * 64 + k0);
      async_g2l_b128(&s_b[pb][r * TLD + k0],
                     wproj + (long)(nt * 64 + r) * D_MODEL + kt * 64 + k0);
    }
  };

  v8f acc0 = {}, acc1 = {};
  int p = 0;
  stage(0, 0);
  for (int kt = 0; kt < NKT; ++kt) {
    __syncthreads();
    bool pre = (kt + 1) < NKT;
    if (pre) stage(kt + 1, p ^ 1);
    if (pre) async_wait_4(); else async_wait_0();
    __syncthreads();
    const __bf16* pa = s_a[p];
    const __bf16* pb = s_b[p];
    v16bf b0  = load_b_frag_nk(pb, TLD, ni * 16, 0, lane);
    v16bf b1  = load_b_frag_nk(pb, TLD, ni * 16, 32, lane);
    v16bf a00 = load_a_frag(pa, TLD, mi0 * 16, 0, lane);
    v16bf a01 = load_a_frag(pa, TLD, mi0 * 16, 32, lane);
    v16bf a10 = load_a_frag(pa, TLD, (mi0 + 2) * 16, 0, lane);
    v16bf a11 = load_a_frag(pa, TLD, (mi0 + 2) * 16, 32, lane);
    acc0 = wmma_bf16(a00, b0, acc0);
    acc0 = wmma_bf16(a01, b1, acc0);
    acc1 = wmma_bf16(a10, b0, acc1);
    acc1 = wmma_bf16(a11, b1, acc1);
    p ^= 1;
  }

  const int n_g = nt * 64 + ni * 16 + (lane & 15);
#pragma unroll
  for (int j = 0; j < 8; ++j) {
    int m_g0 = mt * 64 + mi0 * 16 + j + 8 * (lane >> 4);
    out[(long)m_g0 * D_MODEL + n_g] = acc0[j];
    out[(long)(m_g0 + 32) * D_MODEL + n_g] = acc1[j];
  }
}

// ---------------------------------------------------------------------------
// Host launcher
// ---------------------------------------------------------------------------
extern "C" void kernel_launch(void* const* d_in, const int* in_sizes, int n_in,
                              void* d_out, int out_size, void* d_ws, size_t ws_size,
                              hipStream_t stream) {
  (void)in_sizes; (void)n_in; (void)out_size; (void)ws_size;
  const float* x      = (const float*)d_in[0];   // [B,T,D]
  const float* w_qkv  = (const float*)d_in[1];   // [3D,D]
  const float* w_ky   = (const float*)d_in[2];   // [D,D]
  const float* w_proj = (const float*)d_in[3];   // [D,D]
  float* out = (float*)d_out;

  const size_t X_E    = (size_t)M_TOTAL * D_MODEL;          // 4,194,304
  const size_t WQKV_E = (size_t)3 * D_MODEL * D_MODEL;      // 3,145,728
  const size_t WSQ_E  = (size_t)D_MODEL * D_MODEL;          // 1,048,576
  const size_t HB_E   = (size_t)BATCH * N_HEADS * SEQ * HEAD_DIM;  // 4,194,304

  char* ws = (char*)d_ws;
  size_t off = 0;
  auto carve = [&](size_t elems) { __bf16* p = (__bf16*)(ws + off);
                                   off += ((elems * sizeof(__bf16) + 255) & ~(size_t)255);
                                   return p; };
  __bf16* x_bf     = carve(X_E);
  __bf16* wqkv_bf  = carve(WQKV_E);
  __bf16* wky_bf   = carve(WSQ_E);
  __bf16* wproj_bf = carve(WSQ_E);
  __bf16* q_bf     = carve(HB_E);
  __bf16* k_bf     = carve(HB_E);
  __bf16* vt_bf    = carve(HB_E);   // transposed value layout [B,H,hd,T]
  __bf16* ky_bf    = carve(HB_E);
  __bf16* attn_bf  = carve(X_E);

  // 1) f32 -> bf16
  cvt_f32_bf16<<<(int)(X_E    / 256), 256, 0, stream>>>(x,      x_bf,     (int)X_E);
  cvt_f32_bf16<<<(int)(WQKV_E / 256), 256, 0, stream>>>(w_qkv,  wqkv_bf,  (int)WQKV_E);
  cvt_f32_bf16<<<(int)(WSQ_E  / 256), 256, 0, stream>>>(w_ky,   wky_bf,   (int)WSQ_E);
  cvt_f32_bf16<<<(int)(WSQ_E  / 256), 256, 0, stream>>>(w_proj, wproj_bf, (int)WSQ_E);

  // 2) fused qkv + k_y projections (virtual N = 4096)
  gemm_qkv_ky<<<dim3(M_TOTAL / 64, 4096 / 64), 256, 0, stream>>>(
      x_bf, wqkv_bf, wky_bf, q_bf, k_bf, vt_bf, ky_bf);

  // 3) fused eml attention: grid (q-blocks, heads, batch)
  attn_eml<<<dim3(SEQ / 32, N_HEADS, BATCH), 256, 0, stream>>>(
      q_bf, k_bf, vt_bf, ky_bf, attn_bf);

  // 4) output projection -> f32 d_out
  gemm_proj<<<dim3(M_TOTAL / 64, D_MODEL / 64), 256, 0, stream>>>(
      attn_bf, wproj_bf, out);
}
